// TermQAModel_75565654606341
// MI455X (gfx1250) — matrix operand order repaired and checked by reading, not compiled, not run on hardware
//
#include <hip/hip_runtime.h>
#include <hip/hip_bf16.h>

// Problem dims (fixed by the reference)
#define QD 32
#define TD 64
#define HD 768
#define AD 2048
#define ND 8
#define MDIM (QD * TD)   // 2048 rows   (query terms)
#define NDIM (AD * ND)   // 16384 cols  (candidate heads)

typedef __attribute__((ext_vector_type(16))) __bf16 v16bf;
typedef __attribute__((ext_vector_type(8)))  __bf16 v8bf;
typedef __attribute__((ext_vector_type(8)))  float  v8f;
typedef __attribute__((ext_vector_type(4)))  unsigned int v4u;
typedef __attribute__((ext_vector_type(8)))  unsigned int v8u;

// LDS layout for one staged B tile: 64 rows x 32 bf16, TDM-padded:
// 16 dwords (64B) of data then 4 dwords (16B) pad -> 80B row stride.
#define BROW_LDS   80
#define BTILE_LDS  (64 * BROW_LDS)     // 5120 B per (hi|lo) tile
#define BBUF_LDS   (2 * BTILE_LDS)     // hi + lo = 10240 B per buffer
#define SMEM_BYTES (2 * BBUF_LDS)      // double buffered = 20480 B

// fp32 -> bf16 hi/lo split (x ~= hi + lo), enabling bf16x3 fp32-grade GEMM.
__global__ void split_bf16_kernel(const float* __restrict__ src,
                                  __bf16* __restrict__ hi,
                                  __bf16* __restrict__ lo, int n) {
  int i = blockIdx.x * blockDim.x + threadIdx.x;
  int stride = gridDim.x * blockDim.x;
  for (; i < n; i += stride) {
    float x = src[i];
    __bf16 h = (__bf16)x;
    float r = x - (float)h;
    hi[i] = h;
    lo[i] = (__bf16)r;
  }
}

__global__ void zero_kernel(float* __restrict__ p, int n) {
  int i = blockIdx.x * blockDim.x + threadIdx.x;
  if (i < n) p[i] = 0.0f;
}

// Issue one TDM transfer: a 64-row x 16-dword tile (rows = c-feature vectors,
// row stride 384 dwords in memory) into LDS at lds_addr, padded 4 dwords per
// 16 dwords so the LDS row stride is 80B (bank-conflict-free fragment reads).
// D# per CDNA5 ISA ch.8: group0 = {flags, lds_addr, global_addr, type=2},
// group1 = {modes, tensor dims/strides, tile dims}.
__device__ __forceinline__ void tdm_load_tile(unsigned lds_addr,
                                              const __bf16* gptr) {
  unsigned long long ga = (unsigned long long)(uintptr_t)gptr;
  v4u g0;
  g0[0] = 1u;                                   // count=1, user descriptor
  g0[1] = lds_addr;                             // LDS byte address
  g0[2] = (unsigned)(ga & 0xFFFFFFFFu);         // global_addr[31:0]
  g0[3] = (unsigned)((ga >> 32) & 0x01FFFFFFu)  // global_addr[56:32]
          | (2u << 30);                         // type=2 ("image")
  v8u g1;
  g1[0] = (2u << 16)      // data_size = 4B (dword)
        | (1u << 20)      // pad_enable
        | (3u << 22)      // pad_interval: 16 dwords
        | (3u << 25);     // pad_amount:   4 dwords
  g1[1] = (16u << 16);    // tensor_dim0 = 16 dwords (tile-exact, no OOB)
  g1[2] = (64u << 16);    // tensor_dim0 hi | tensor_dim1 = 64 rows
  g1[3] = (16u << 16);    // tensor_dim1 hi | tile_dim0 = 16 dwords
  g1[4] = 64u;            // tile_dim1 = 64 rows, tile_dim2 = 0 (2D)
  g1[5] = 384u;           // tensor_dim0_stride = 384 dwords (= H)
  g1[6] = 0u;             // stride hi | tensor_dim1_stride (unused, 2D)
  g1[7] = 0u;
  asm volatile("tensor_load_to_lds %0, %1" :: "s"(g0), "s"(g1) : "memory");
}

// Block = 8 waves sharing one 64-column B strip (TDM-staged in LDS, double
// buffered); wave w owns M-tile (blockIdx.y*8 + w). bf16x3 split GEMM with
// v_wmma_f32_16x16x32_bf16, softmax-expectation epilogue in registers.
__launch_bounds__(256)
__global__ void termqa_wmma_kernel(const __bf16* __restrict__ qhi,
                                   const __bf16* __restrict__ qlo,
                                   const __bf16* __restrict__ chi,
                                   const __bf16* __restrict__ clo,
                                   float* __restrict__ out) {
  extern __shared__ unsigned char smem[];   // SMEM_BYTES
  const int lane = threadIdx.x & 31;
  const int wave = threadIdx.x >> 5;
  const int half = lane >> 4;               // 0: lanes 0-15, 1: lanes 16-31
  const int l15  = lane & 15;

  const int mtile = blockIdx.y * 8 + wave;  // 16 rows per wave
  const int nbase = blockIdx.x * 64;        // 64 cols per block (shared)

  // A fragment source: q row for this lane (row-major, K contiguous)
  const int mrow = mtile * 16 + l15;
  const __bf16* aH = qhi + (size_t)mrow * HD;
  const __bf16* aL = qlo + (size_t)mrow * HD;

  // B strip base (rows of c == columns of B), staged by the TDM
  const __bf16* bHg = chi + (size_t)nbase * HD;
  const __bf16* bLg = clo + (size_t)nbase * HD;
  const unsigned lds_base = (unsigned)(uintptr_t)(&smem[0]); // low 32 = LDS off

  v8f acc[4] = {v8f{}, v8f{}, v8f{}, v8f{}};

  // Prologue: stage K-chunk 0 into buffer 0
  if (wave == 0) {
    tdm_load_tile(lds_base,             bHg);
    tdm_load_tile(lds_base + BTILE_LDS, bLg);
  }

  int cur = 0;
  for (int kt = 0; kt < HD / 32; ++kt) {
    const int kk = kt * 32;
    if (wave == 0) __builtin_amdgcn_s_wait_tensorcnt(0);
    __syncthreads();                       // buffer `cur` visible to all waves
    if (wave == 0 && kt + 1 < HD / 32) {   // prefetch next chunk into cur^1
      const unsigned nb = lds_base + (unsigned)(cur ^ 1) * BBUF_LDS;
      tdm_load_tile(nb,             bHg + kk + 32);
      tdm_load_tile(nb + BTILE_LDS, bLg + kk + 32);
    }

    // A fragment (contiguous K order, matching B): lanes 0-15 hold K[kk+0..15];
    // lanes 16-31 hold K[kk+16..31].
    v16bf ahi, alo;
    {
      const v8bf p0 = *(const v8bf*)(aH + kk + half * 16);
      const v8bf p1 = *(const v8bf*)(aH + kk + half * 16 + 8);
      const v8bf q0 = *(const v8bf*)(aL + kk + half * 16);
      const v8bf q1 = *(const v8bf*)(aL + kk + half * 16 + 8);
#pragma unroll
      for (int i = 0; i < 8; ++i) {
        ahi[i] = p0[i]; ahi[8 + i] = p1[i];
        alo[i] = q0[i]; alo[8 + i] = q1[i];
      }
    }

    const unsigned char* bufh = smem + cur * BBUF_LDS;
    const unsigned char* bufl = bufh + BTILE_LDS;
#pragma unroll
    for (int j = 0; j < 4; ++j) {
      // B 32x16 fragment from LDS: lane's column = tile row j*16+l15,
      // 16 contiguous K starting at half*16 -> two 16B ds loads (hi + lo).
      const int boff = (j * 16 + l15) * BROW_LDS + half * 32;
      const v8bf r0 = *(const v8bf*)(bufh + boff);
      const v8bf r1 = *(const v8bf*)(bufh + boff + 16);
      const v8bf s0 = *(const v8bf*)(bufl + boff);
      const v8bf s1 = *(const v8bf*)(bufl + boff + 16);
      v16bf bhi, blo;
#pragma unroll
      for (int i = 0; i < 8; ++i) {
        bhi[i] = r0[i]; bhi[8 + i] = r1[i];
        blo[i] = s0[i]; blo[8 + i] = s1[i];
      }
      // bf16x3: hi*hi + hi*lo + lo*hi, f32 accumulate
      acc[j] = __builtin_amdgcn_wmma_f32_16x16x32_bf16(
          false, ahi, false, bhi, (short)0, acc[j], false, false);
      acc[j] = __builtin_amdgcn_wmma_f32_16x16x32_bf16(
          false, ahi, false, blo, (short)0, acc[j], false, false);
      acc[j] = __builtin_amdgcn_wmma_f32_16x16x32_bf16(
          false, alo, false, bhi, (short)0, acc[j], false, false);
    }
    __syncthreads();                       // reads of `cur` done before reuse
    cur ^= 1;
  }

  // Epilogue. C/D layout: reg r on lane l holds logits[M = r + 8*half, N = l15].
  // Softmax groups (8 heads of one doc) = 8 consecutive lanes -> xor 1/2/4
  // butterfly. Rows fold: in-lane sum over r (8 rows) + xor 16 (other 8 rows).
  const int q = mtile >> 2;   // 4 row-tiles (of 16) per q since T = 64
#pragma unroll
  for (int j = 0; j < 4; ++j) {
    float s = 0.0f;
#pragma unroll
    for (int r = 0; r < 8; ++r) {
      float v = acc[j][r];
      float m = v;
      m = fmaxf(m, __shfl_xor(m, 1));
      m = fmaxf(m, __shfl_xor(m, 2));
      m = fmaxf(m, __shfl_xor(m, 4));
      float p   = __expf(v - m);
      float num = p * v;
      float den = p;
      num += __shfl_xor(num, 1); den += __shfl_xor(den, 1);
      num += __shfl_xor(num, 2); den += __shfl_xor(den, 2);
      num += __shfl_xor(num, 4); den += __shfl_xor(den, 4);
      float match = num / den;              // sum_n p_n * logit_n
      s += __logf(fmaxf(match, 0.0f) + 1.0f);
    }
    s += __shfl_xor(s, 16);                 // fold the other 8 rows
    if (lane < 16 && (lane & 7) == 0) {     // one lane per doc group
      int doc = (nbase + j * 16 + lane) >> 3;
      atomicAdd(out + q * AD + doc, s);
    }
  }
}

extern "C" void kernel_launch(void* const* d_in, const int* in_sizes, int n_in,
                              void* d_out, int out_size, void* d_ws, size_t ws_size,
                              hipStream_t stream) {
  const float* qf = (const float*)d_in[0];   // [32, 64, 768] fp32
  const float* cf = (const float*)d_in[1];   // [2048, 8, 768] fp32
  float* out = (float*)d_out;                // [32, 2048] fp32

  const size_t qn = (size_t)QD * TD * HD;    // 1,572,864
  const size_t cn = (size_t)AD * ND * HD;    // 12,582,912

  __bf16* qhi = (__bf16*)d_ws;
  __bf16* qlo = qhi + qn;
  __bf16* chi = qlo + qn;
  __bf16* clo = chi + cn;                    // total ~56.6 MB of d_ws

  split_bf16_kernel<<<2048, 256, 0, stream>>>(qf, qhi, qlo, (int)qn);
  split_bf16_kernel<<<8192, 256, 0, stream>>>(cf, chi, clo, (int)cn);
  zero_kernel<<<(QD * AD + 255) / 256, 256, 0, stream>>>(out, QD * AD);

  dim3 grid(NDIM / 64, MDIM / 16 / 8);       // (256, 16): strip x mtile-group
  termqa_wmma_kernel<<<grid, 256, SMEM_BYTES, stream>>>(qhi, qlo, chi, clo, out);
}
